// NSABlock_67765993997286
// MI455X (gfx1250) — matrix operand order repaired and checked by reading, not compiled, not run on hardware
//
#include <hip/hip_runtime.h>
#include <hip/hip_bf16.h>

typedef __bf16 bf16;
typedef __attribute__((ext_vector_type(8)))  bf16  v8bf;
typedef __attribute__((ext_vector_type(16))) bf16  v16bf;
typedef __attribute__((ext_vector_type(8)))  float v8f;

#define DIM_  256
#define NHEAD 8
#define HD    32
#define KS_   7
#define BB    4
#define HH_   48
#define WW_   48
#define SS    (HH_ * WW_)      /* 2304 tokens per batch */
#define NT    (BB * SS)        /* 9216 total tokens */
#define EE    4
#define HID_  1024
#define KTOK  576              /* tokens per expert */

__device__ __forceinline__ bf16 f2bf(float f) {
  unsigned u = __builtin_bit_cast(unsigned, f);
  unsigned r = (u + 0x7FFFu + ((u >> 16) & 1u)) >> 16;
  return __builtin_bit_cast(bf16, (unsigned short)r);
}

// ---------------------------------------------------------------- weights
// dst[e][n][k] = src[e][k][n]  (bf16, row-major [N][K] per batch e)
__global__ void k_transpose_bf16(const float* __restrict__ src, bf16* __restrict__ dst,
                                 int K, int N, int total) {
  int i = blockIdx.x * 256 + threadIdx.x;
  if (i >= total) return;
  int per = K * N;
  int e = i / per, r = i - e * per;
  int k = r / N, n = r - k * N;
  dst[(size_t)e * per + (size_t)n * K + k] = f2bf(src[i]);
}

// ---------------------------------------------------------------- layernorm 1
__global__ void k_ln1(const float* __restrict__ x, const float* __restrict__ g,
                      const float* __restrict__ bt, bf16* __restrict__ xa) {
  __shared__ float r1[256], r2[256];
  int t = blockIdx.x, d = threadIdx.x;
  float v = x[(size_t)t * DIM_ + d];
  r1[d] = v; r2[d] = v * v;
  __syncthreads();
  for (int s = 128; s > 0; s >>= 1) {
    if (d < s) { r1[d] += r1[d + s]; r2[d] += r2[d + s]; }
    __syncthreads();
  }
  float m   = r1[0] * (1.0f / DIM_);
  float var = r2[0] * (1.0f / DIM_) - m * m;
  xa[(size_t)t * DIM_ + d] = f2bf((v - m) * rsqrtf(var + 1e-5f) * g[d] + bt[d]);
}

// ---------------------------------------------------------------- WMMA fragments
// A fragment per ISA 16-bit 16x32 layout: elems 0-7 = K h*8+{0..7}, 8-15 = K 16+h*8+{0..7}
__device__ __forceinline__ v16bf load_afrag(const bf16* __restrict__ ap, int k0, int h) {
  v8bf a0 = *(const v8bf*)(ap + k0 + h * 8);
  v8bf a1 = *(const v8bf*)(ap + k0 + 16 + h * 8);
  return __builtin_shufflevector(a0, a1, 0,1,2,3,4,5,6,7,8,9,10,11,12,13,14,15);
}
// B fragment (B^T row-major): elems 0-7 = K h*8+{0..7}, 8-15 = K 16+h*8+{0..7}
__device__ __forceinline__ v16bf load_bfrag(const bf16* __restrict__ bp, int k0, int h) {
  v8bf b0 = *(const v8bf*)(bp + k0 + h * 8);
  v8bf b1 = *(const v8bf*)(bp + k0 + 16 + h * 8);
  return __builtin_shufflevector(b0, b1, 0,1,2,3,4,5,6,7,8,9,10,11,12,13,14,15);
}

// Wave computes a 64x64 output tile: 4x4 16x16 fragments, 16 WMMAs per 32-K step.
// Per step: 4 KB A + 4 KB B operands for 256 KFLOP -> 32 FLOP/byte from cache.
// K is compile-time so fragment strides fold into load-immediate offsets:
// only one A base + one B base pointer per lane (gather path keeps 4 A pointers).
template<int K, bool GATHER>
__device__ __forceinline__ void wmma_block64(v8f acc[4][4],
    const bf16* __restrict__ A, const bf16* __restrict__ BT,
    int m0, int n0, const int* __restrict__ rows, int rowbase) {
  int lane = threadIdx.x & 31, l16 = lane & 15, h = lane >> 4;
  const bf16* ap0 = A + (size_t)(m0 + l16) * K;
  const bf16* apg[4];
  if (GATHER) {
#pragma unroll
    for (int f = 0; f < 4; ++f)
      apg[f] = A + (size_t)(rowbase + rows[m0 + 16 * f + l16]) * K;
  }
  const bf16* bp0 = BT + (size_t)(n0 + l16) * K;
#pragma unroll
  for (int f = 0; f < 4; ++f)
#pragma unroll
    for (int g = 0; g < 4; ++g)
      acc[f][g] = (v8f){0.f, 0.f, 0.f, 0.f, 0.f, 0.f, 0.f, 0.f};
  for (int k0 = 0; k0 < K; k0 += 32) {
    v16bf a[4], b[4];
#pragma unroll
    for (int f = 0; f < 4; ++f)
      a[f] = load_afrag(GATHER ? apg[f] : (ap0 + f * (16 * K)), k0, h);
#pragma unroll
    for (int g = 0; g < 4; ++g)
      b[g] = load_bfrag(bp0 + g * (16 * K), k0, h);
#pragma unroll
    for (int f = 0; f < 4; ++f)
#pragma unroll
      for (int g = 0; g < 4; ++g)
        acc[f][g] = __builtin_amdgcn_wmma_f32_16x16x32_bf16(
            false, a[f], false, b[g], (short)0, acc[f][g], false, false);
  }
}

// ---------------------------------------------------------------- C = A*B^T + bias
template<int K>
__global__ __launch_bounds__(256, 1)
void k_gemm_bias(const bf16* __restrict__ A, const bf16* __restrict__ BT,
                 const float* __restrict__ bias, float* __restrict__ C,
                 int M, int N) {
  int wave = blockIdx.x * 8 + (threadIdx.x >> 5);
  int ntn = N >> 6;
  int tm = wave / ntn, tn = wave - tm * ntn;
  if (tm >= (M >> 6)) return;                       // wave-uniform
  int lane = threadIdx.x & 31, l16 = lane & 15, h = lane >> 4;
  int m0 = tm << 6, n0 = tn << 6;
  v8f acc[4][4];
  wmma_block64<K, false>(acc, A, BT, m0, n0, nullptr, 0);
#pragma unroll
  for (int g = 0; g < 4; ++g) {
    int col = n0 + 16 * g + l16;
    float bv = bias[col];
#pragma unroll
    for (int f = 0; f < 4; ++f)
#pragma unroll
      for (int r = 0; r < 8; ++r)
        C[(size_t)(m0 + 16 * f + r + 8 * h) * N + col] = acc[f][g][r] + bv;
  }
}

// ---------------------------------------------------------------- C = resid + A*B^T + bias
template<int K>
__global__ __launch_bounds__(256, 1)
void k_gemm_proj(const bf16* __restrict__ A, const bf16* __restrict__ BT,
                 const float* __restrict__ bias, const float* __restrict__ resid,
                 float* __restrict__ C, int M, int N) {
  int wave = blockIdx.x * 8 + (threadIdx.x >> 5);
  int ntn = N >> 6;
  int tm = wave / ntn, tn = wave - tm * ntn;
  if (tm >= (M >> 6)) return;
  int lane = threadIdx.x & 31, l16 = lane & 15, h = lane >> 4;
  int m0 = tm << 6, n0 = tn << 6;
  v8f acc[4][4];
  wmma_block64<K, false>(acc, A, BT, m0, n0, nullptr, 0);
#pragma unroll
  for (int g = 0; g < 4; ++g) {
    int col = n0 + 16 * g + l16;
    float bv = bias[col];
#pragma unroll
    for (int f = 0; f < 4; ++f)
#pragma unroll
      for (int r = 0; r < 8; ++r) {
        size_t o = (size_t)(m0 + 16 * f + r + 8 * h) * N + col;
        C[o] = resid[o] + acc[f][g][r] + bv;
      }
  }
}

// ---------------------------------------------------------------- neighborhood attention
// one thread per (b, head, i, j); online softmax, all state in registers
__global__ void k_attn(const float* __restrict__ qkv, const float* __restrict__ rpb,
                       bf16* __restrict__ ao) {
  int gid = blockIdx.x * 256 + threadIdx.x;
  int j = gid % WW_; int r_ = gid / WW_;
  int i = r_ % HH_;  r_ /= HH_;
  int hh = r_ % NHEAD; int b = r_ / NHEAD;
  int tok = (b * HH_ + i) * WW_ + j;
  const float* qp = qkv + (size_t)tok * (3 * DIM_) + hh * HD;
  float q[HD];
#pragma unroll
  for (int d = 0; d < HD; ++d) q[d] = qp[d] * 0.17677669529663687f;  // hd^-0.5
  int sti = i - 3; if (sti < 0) sti = 0; if (sti > HH_ - KS_) sti = HH_ - KS_;
  int stj = j - 3; if (stj < 0) stj = 0; if (stj > WW_ - KS_) stj = WW_ - KS_;
  float m = -3.4e38f, l = 0.f;
  float o[HD];
#pragma unroll
  for (int d = 0; d < HD; ++d) o[d] = 0.f;
  for (int a = 0; a < KS_; ++a) {
    for (int c = 0; c < KS_; ++c) {
      int nt = (b * HH_ + sti + a) * WW_ + stj + c;
      const float* kp = qkv + (size_t)nt * (3 * DIM_) + DIM_ + hh * HD;
      const float* vp = kp + DIM_;
      float s = 0.f;
#pragma unroll
      for (int d = 0; d < HD; ++d) s += q[d] * kp[d];
      s += rpb[hh * 169 + (sti + a - i + 6) * 13 + (stj + c - j + 6)];
      float nm   = fmaxf(m, s);
      float corr = __expf(m - nm);
      float p    = __expf(s - nm);
      l = l * corr + p;
#pragma unroll
      for (int d = 0; d < HD; ++d) o[d] = o[d] * corr + p * vp[d];
      m = nm;
    }
  }
  float inv = 1.f / l;
#pragma unroll
  for (int d = 0; d < HD; ++d)
    ao[(size_t)tok * DIM_ + hh * HD + d] = f2bf(o[d] * inv);
}

// ---------------------------------------------------------------- layernorm 2 + gate softmax
__global__ void k_ln2_gate(const float* __restrict__ xf, const float* __restrict__ g,
                           const float* __restrict__ bt, const float* __restrict__ gw,
                           bf16* __restrict__ xn, float* __restrict__ affT) {
  __shared__ float r1[256], r2[256];
  __shared__ float logits[EE];
  int t = blockIdx.x, d = threadIdx.x;
  float v = xf[(size_t)t * DIM_ + d];
  r1[d] = v; r2[d] = v * v;
  __syncthreads();
  for (int s = 128; s > 0; s >>= 1) {
    if (d < s) { r1[d] += r1[d + s]; r2[d] += r2[d + s]; }
    __syncthreads();
  }
  float m   = r1[0] * (1.0f / DIM_);
  float var = r2[0] * (1.0f / DIM_) - m * m;
  float ny  = (v - m) * rsqrtf(var + 1e-5f) * g[d] + bt[d];
  xn[(size_t)t * DIM_ + d] = f2bf(ny);
  for (int e = 0; e < EE; ++e) {
    __syncthreads();
    r1[d] = ny * gw[d * EE + e];
    __syncthreads();
    for (int s = 128; s > 0; s >>= 1) {
      if (d < s) r1[d] += r1[d + s];
      __syncthreads();
    }
    if (d == 0) logits[e] = r1[0];
  }
  __syncthreads();
  if (d == 0) {
    float mx = fmaxf(fmaxf(logits[0], logits[1]), fmaxf(logits[2], logits[3]));
    float ex[EE], sum = 0.f;
    for (int e = 0; e < EE; ++e) { ex[e] = __expf(logits[e] - mx); sum += ex[e]; }
    int b = t / SS, s = t - b * SS;
    for (int e = 0; e < EE; ++e)
      affT[(size_t)(b * EE + e) * SS + s] = ex[e] / sum;
  }
}

// ---------------------------------------------------------------- exact top-576 per (b,e)
// bitonic sort of 4096 packed keys in LDS; ascending key == descending value,
// ties broken toward lower index (matches jax.lax.top_k)
__global__ void k_topk(const float* __restrict__ affT, int* __restrict__ idx,
                       float* __restrict__ gate) {
  __shared__ unsigned long long sk[4096];
  int be = blockIdx.x, tid = threadIdx.x;
  const float* a = affT + (size_t)be * SS;
  for (int i = tid; i < 4096; i += 256) {
    unsigned long long key = ~0ull;
    if (i < SS) {
      unsigned u  = __builtin_bit_cast(unsigned, a[i]);
      unsigned su = (u & 0x80000000u) ? ~u : (u | 0x80000000u);  // monotone map
      key = ((unsigned long long)(~su) << 32) | (unsigned)i;
    }
    sk[i] = key;
  }
  for (unsigned sz = 2; sz <= 4096; sz <<= 1)
    for (unsigned j = sz >> 1; j > 0; j >>= 1) {
      __syncthreads();
      for (unsigned i = tid; i < 4096; i += 256) {
        unsigned ixj = i ^ j;
        if (ixj > i) {
          unsigned long long x = sk[i], y = sk[ixj];
          bool up = ((i & sz) == 0);
          if (up ? (x > y) : (x < y)) { sk[i] = y; sk[ixj] = x; }
        }
      }
    }
  __syncthreads();
  for (int r = tid; r < KTOK; r += 256) {
    int s = (int)(sk[r] & 0xFFFFFFFFull);
    idx[(size_t)be * KTOK + r]  = s;
    gate[(size_t)be * KTOK + r] = a[s];
  }
}

// ---------------------------------------------------------------- MoE FC1: gather + GEMM + GELU
__global__ __launch_bounds__(256, 1)
void k_moe_mlp1(const bf16* __restrict__ xn, const bf16* __restrict__ w1T,
                const float* __restrict__ b1, const int* __restrict__ idx,
                bf16* __restrict__ hdn) {
  int wave = blockIdx.x * 8 + (threadIdx.x >> 5);
  int be = wave / (9 * 16);                  // (b,e) pair; tiles: 576/64 x 1024/64
  int t  = wave - be * (9 * 16);
  int tm = t >> 4, tn = t & 15;
  int e = be & 3, b = be >> 2;
  int lane = threadIdx.x & 31, l16 = lane & 15, h = lane >> 4;
  int m0 = tm << 6, n0 = tn << 6;
  v8f acc[4][4];
  wmma_block64<DIM_, true>(acc, xn, w1T + (size_t)e * DIM_ * HID_,
                           m0, n0, idx + (size_t)be * KTOK, b * SS);
  bf16* out = hdn + (size_t)be * KTOK * HID_;
#pragma unroll
  for (int g = 0; g < 4; ++g) {
    int col = n0 + 16 * g + l16;
    float bv = b1[e * HID_ + col];
#pragma unroll
    for (int f = 0; f < 4; ++f)
#pragma unroll
      for (int r = 0; r < 8; ++r) {
        float v = acc[f][g][r] + bv;
        float gel = 0.5f * v * (1.f + erff(v * 0.7071067811865475f));  // exact GELU
        out[(size_t)(m0 + 16 * f + r + 8 * h) * HID_ + col] = f2bf(gel);
      }
  }
}

// ---------------------------------------------------------------- MoE FC2: GEMM + gate + scatter-add
__global__ __launch_bounds__(256, 1)
void k_moe_mlp2(const bf16* __restrict__ hdn, const bf16* __restrict__ w2T,
                const float* __restrict__ b2, const int* __restrict__ idx,
                const float* __restrict__ gate, float* __restrict__ out) {
  int wave = blockIdx.x * 8 + (threadIdx.x >> 5);
  int be = wave / (9 * 4);                   // tiles: 576/64 x 256/64
  int t  = wave - be * (9 * 4);
  int tm = t >> 2, tn = t & 3;
  int e = be & 3, b = be >> 2;
  int lane = threadIdx.x & 31, l16 = lane & 15, h = lane >> 4;
  int m0 = tm << 6, n0 = tn << 6;
  v8f acc[4][4];
  wmma_block64<HID_, false>(acc, hdn + (size_t)be * KTOK * HID_,
                            w2T + (size_t)e * DIM_ * HID_, m0, n0, nullptr, 0);
#pragma unroll
  for (int g = 0; g < 4; ++g) {
    int col = n0 + 16 * g + l16;
    float bv = b2[e * DIM_ + col];
#pragma unroll
    for (int f = 0; f < 4; ++f)
#pragma unroll
      for (int r = 0; r < 8; ++r) {
        int m   = m0 + 16 * f + r + 8 * h;
        int tok = b * SS + idx[(size_t)be * KTOK + m];
        float gv = gate[(size_t)be * KTOK + m];
        atomicAdd(out + (size_t)tok * DIM_ + col,
                  (acc[f][g][r] + bv) * gv);           // global_atomic_add_f32
      }
  }
}

// ================================================================ launch
extern "C" void kernel_launch(void* const* d_in, const int* in_sizes, int n_in,
                              void* d_out, int out_size, void* d_ws, size_t ws_size,
                              hipStream_t stream) {
  (void)in_sizes; (void)n_in; (void)out_size; (void)ws_size;
  const float* x      = (const float*)d_in[0];
  const float* ln1_g  = (const float*)d_in[1];
  const float* ln1_b  = (const float*)d_in[2];
  const float* qkv_w  = (const float*)d_in[3];
  const float* qkv_b  = (const float*)d_in[4];
  const float* rpb    = (const float*)d_in[5];
  const float* proj_w = (const float*)d_in[6];
  const float* proj_b = (const float*)d_in[7];
  const float* ln2_g  = (const float*)d_in[8];
  const float* ln2_b  = (const float*)d_in[9];
  const float* gate_w = (const float*)d_in[10];
  const float* w1     = (const float*)d_in[11];
  const float* b1     = (const float*)d_in[12];
  const float* w2     = (const float*)d_in[13];
  const float* b2     = (const float*)d_in[14];
  float* out = (float*)d_out;

  char* ws = (char*)d_ws;
  size_t off = 0;
  auto alloc = [&](size_t bytes) -> char* {
    char* p = ws + off;
    off += (bytes + 255) & ~(size_t)255;
    return p;
  };
  bf16*  qkvT  = (bf16*)alloc((size_t)768 * 256 * 2);
  bf16*  projT = (bf16*)alloc((size_t)256 * 256 * 2);
  bf16*  w1T   = (bf16*)alloc((size_t)EE * 256 * 1024 * 2);
  bf16*  w2T   = (bf16*)alloc((size_t)EE * 1024 * 256 * 2);
  bf16*  xa    = (bf16*)alloc((size_t)NT * DIM_ * 2);
  float* qkv   = (float*)alloc((size_t)NT * 768 * 4);
  bf16*  aout  = (bf16*)alloc((size_t)NT * DIM_ * 2);
  bf16*  xn    = (bf16*)alloc((size_t)NT * DIM_ * 2);
  float* affT  = (float*)alloc((size_t)BB * EE * SS * 4);
  int*   idx   = (int*)  alloc((size_t)BB * EE * KTOK * 4);
  float* gate  = (float*)alloc((size_t)BB * EE * KTOK * 4);
  bf16*  hdn   = (bf16*)alloc((size_t)BB * EE * KTOK * HID_ * 2);

  // weight prep (bf16, transposed to [N][K])
  k_transpose_bf16<<<768, 256, 0, stream>>>(qkv_w, qkvT, 256, 768, 256 * 768);
  k_transpose_bf16<<<256, 256, 0, stream>>>(proj_w, projT, 256, 256, 256 * 256);
  k_transpose_bf16<<<4096, 256, 0, stream>>>(w1, w1T, 256, 1024, EE * 256 * 1024);
  k_transpose_bf16<<<4096, 256, 0, stream>>>(w2, w2T, 1024, 256, EE * 1024 * 256);

  // attention branch
  k_ln1<<<NT, 256, 0, stream>>>(x, ln1_g, ln1_b, xa);
  k_gemm_bias<256><<<(NT / 64) * (768 / 64) / 8, 256, 0, stream>>>(xa, qkvT, qkv_b, qkv,
                                                                   NT, 768);
  k_attn<<<(BB * NHEAD * HH_ * WW_) / 256, 256, 0, stream>>>(qkv, rpb, aout);
  k_gemm_proj<256><<<(NT / 64) * (256 / 64) / 8, 256, 0, stream>>>(aout, projT, proj_b,
                                                                   x, out, NT, 256);

  // MoE branch
  k_ln2_gate<<<NT, 256, 0, stream>>>(out, ln2_g, ln2_b, gate_w, xn, affT);
  k_topk<<<BB * EE, 256, 0, stream>>>(affT, idx, gate);
  k_moe_mlp1<<<(BB * EE * 9 * 16) / 8, 256, 0, stream>>>(xn, w1T, b1, idx, hdn);
  k_moe_mlp2<<<(BB * EE * 9 * 4) / 8, 256, 0, stream>>>(hdn, w2T, b2, idx, gate, out);
}
